// SelfAttention_2920577761623
// MI455X (gfx1250) — matrix operand order repaired and checked by reading, not compiled, not run on hardware
//
#include <hip/hip_runtime.h>
#include <cstdint>

typedef __attribute__((ext_vector_type(16))) __bf16 v16bf;
typedef __attribute__((ext_vector_type(8)))  __bf16 v8bf;
typedef __attribute__((ext_vector_type(4)))  __bf16 v4bf;
typedef __attribute__((ext_vector_type(8)))  float  v8f;
typedef __attribute__((ext_vector_type(4)))  unsigned int v4u;
typedef __attribute__((ext_vector_type(4)))  int v4i;
typedef __attribute__((ext_vector_type(8)))  int v8i;

#if defined(__gfx1250__) && __has_builtin(__builtin_amdgcn_tensor_load_to_lds)
#define USE_TDM 1
#else
#define USE_TDM 0
#endif

// ---------------------------------------------------------------------------
// LDS fragment loaders matching CDNA5 WMMA 16-bit layouts (ISA 7.12.2).
// A-matrix 16x32: lane m(0-15)=row m holds K=0..7 (VGPR0-3) & K=16..23 (VGPR4-7);
//                 lanes 16-31 hold K=8..15 & 24..31 of the same rows.
// B-matrix 32x16: lane n(0-15)=col n holds K=0..7 & K=16..23; lanes 16-31 hold K=8..15 & 24..31.
// ---------------------------------------------------------------------------
static __device__ inline v16bf load_a_frag_lds(const __bf16* base, int row, int kk, int ld) {
  const int lane = threadIdx.x & 31;
  const __bf16* p = base + (size_t)(row + (lane & 15)) * ld + kk + ((lane >> 4) << 3);
  v16bf a;
  ((v8bf*)&a)[0] = *(const v8bf*)p;
  ((v8bf*)&a)[1] = *(const v8bf*)(p + 16);
  return a;
}
static __device__ inline v16bf load_b_frag_lds(const __bf16* base, int col, int kk, int ld) {
  const int lane = threadIdx.x & 31;
  const __bf16* p = base + (size_t)(col + (lane & 15)) * ld + kk + ((lane >> 4) << 3);
  v16bf b;
  ((v8bf*)&b)[0] = *(const v8bf*)p;
  ((v8bf*)&b)[1] = *(const v8bf*)(p + 16);
  return b;
}

#if USE_TDM
// ---------------------------------------------------------------------------
// Tensor Data Mover: 2D tile load (bf16, data_size=2B) per ISA ch.8 D# layout.
//   group0: [1:0]=count=1, [63:32]=lds_addr, [120:64]=global_addr, [127:126]=2
//   group1: w0=flags (data_size|pad cfg), dims/strides/tile dims
// flags_w0: 0x00010000 = data_size 2B; optionally |pad_enable|interval|amount.
// ---------------------------------------------------------------------------
static __device__ inline unsigned lds_off32(const void* p) {
  return (unsigned)(unsigned long long)(uintptr_t)p;   // low 32b of flat = LDS offset
}
static __device__ inline void tdm_load_2d(unsigned lds_addr, const void* gptr,
                                          unsigned tensor_d0, unsigned tensor_d1,
                                          unsigned stride_elems,
                                          unsigned tile_d0, unsigned tile_d1,
                                          unsigned flags_w0) {
  const unsigned long long ga = (unsigned long long)(uintptr_t)gptr;
  v4u g0 = { 1u,                                   // count=1, user descriptor
             lds_addr,
             (unsigned)(ga & 0xFFFFFFFFu),
             (unsigned)((ga >> 32) & 0x01FFFFFFu) | (2u << 30) };  // type=2
  v8i g1 = { (int)flags_w0,
             (int)((tensor_d0 & 0xFFFFu) << 16),                    // dim0[15:0]
             (int)((tensor_d0 >> 16) | ((tensor_d1 & 0xFFFFu) << 16)),
             (int)((tensor_d1 >> 16) | (tile_d0 << 16)),
             (int)(tile_d1),                                        // tile_dim1 (tile_dim2=0)
             (int)stride_elems,                                     // dim0_stride[31:0]
             0, 0 };
  v4i g2 = { 1, 1, 0, 0 };   // tensor_dim2=1, tensor_dim3=1, stride2=0, tile_dim3=0
  v4i g3 = { 0, 0, 0, 0 };
#if __clang_major__ >= 23
  v8i gz8 = { 0, 0, 0, 0, 0, 0, 0, 0 };
  __builtin_amdgcn_tensor_load_to_lds(g0, g1, g2, g3, gz8, 0);
#else
  __builtin_amdgcn_tensor_load_to_lds(g0, g1, g2, g3, 0);
#endif
}
#endif // USE_TDM

// ---------------------------------------------------------------------------
// fp32 -> bf16 conversion, 4 elements/thread/iter
// ---------------------------------------------------------------------------
__global__ void f32_to_bf16_kernel(const float* __restrict__ src, __bf16* __restrict__ dst, size_t n4) {
  size_t i = (size_t)blockIdx.x * blockDim.x + threadIdx.x;
  const size_t stride = (size_t)gridDim.x * blockDim.x;
  for (; i < n4; i += stride) {
    float4 f = ((const float4*)src)[i];
    v4bf o = { (__bf16)f.x, (__bf16)f.y, (__bf16)f.z, (__bf16)f.w };
    ((v4bf*)dst)[i] = o;
  }
}

// ---------------------------------------------------------------------------
// C[M,N] = (A[M,K] · Bw[N,K]^T) * out_scale (+ bias)   -- torch Linear style.
// 128x128 block tile, 8 waves in a 4x2 grid, each wave 32x64 (2x4 WMMA tiles).
// TDM path: double-buffered LDS panels (K-step 64); wave 0 issues
// tensor_load_to_lds for the next panel while all waves run WMMA on the
// current one (TENSORcnt + barrier handoff). Fallback: manual vector loads.
// ---------------------------------------------------------------------------
template<bool WRITE_F32, bool ADD_BIAS>
__global__ __launch_bounds__(256) void gemm_bf16_wmma(
    const __bf16* __restrict__ A, const __bf16* __restrict__ Bw,
    const float* __restrict__ bias, void* __restrict__ Cout,
    int M, int N, int K, float out_scale)
{
  constexpr int LDT = 64;                       // unpadded: TDM writes dense rows
  __shared__ __bf16 As[2][128 * LDT];
  __shared__ __bf16 Bs[2][128 * LDT];
  const int m0 = blockIdx.x * 128;
  const int n0 = blockIdx.y * 128;
  const int tid = threadIdx.x;
  const int wid = tid >> 5;
  const int wm = (wid & 3) << 5;   // 0,32,64,96
  const int wn = (wid >> 2) << 6;  // 0,64
  v8f acc[2][4];
#pragma unroll
  for (int i = 0; i < 2; ++i)
#pragma unroll
    for (int j = 0; j < 4; ++j) acc[i][j] = v8f{};

#if USE_TDM
  const int nk = K / 64;
  if (wid == 0) {
    tdm_load_2d(lds_off32(&As[0][0]), &A [(size_t)m0 * K], 64, 128, (unsigned)K, 64, 128, 0x00010000u);
    tdm_load_2d(lds_off32(&Bs[0][0]), &Bw[(size_t)n0 * K], 64, 128, (unsigned)K, 64, 128, 0x00010000u);
  }
  int cur = 0;
  for (int kt = 0; kt < nk; ++kt) {
    if (wid == 0) __builtin_amdgcn_s_wait_tensorcnt(0);  // panel `cur` landed
    __syncthreads();
    if (wid == 0 && kt + 1 < nk) {                       // DMA next panel now
      const size_t k0n = (size_t)(kt + 1) * 64;
      tdm_load_2d(lds_off32(&As[cur ^ 1][0]), &A [(size_t)m0 * K + k0n], 64, 128, (unsigned)K, 64, 128, 0x00010000u);
      tdm_load_2d(lds_off32(&Bs[cur ^ 1][0]), &Bw[(size_t)n0 * K + k0n], 64, 128, (unsigned)K, 64, 128, 0x00010000u);
    }
#pragma unroll
    for (int ks = 0; ks < 64; ks += 32) {
      v16bf af[2], bfv[4];
#pragma unroll
      for (int i = 0; i < 2; ++i) af[i]  = load_a_frag_lds(&As[cur][0], wm + i * 16, ks, LDT);
#pragma unroll
      for (int j = 0; j < 4; ++j) bfv[j] = load_b_frag_lds(&Bs[cur][0], wn + j * 16, ks, LDT);
#pragma unroll
      for (int i = 0; i < 2; ++i)
#pragma unroll
        for (int j = 0; j < 4; ++j)
          acc[i][j] = __builtin_amdgcn_wmma_f32_16x16x32_bf16(
              false, af[i], false, bfv[j], (short)0, acc[i][j], false, false);
    }
    cur ^= 1;
  }
#else
  for (int k0 = 0; k0 < K; k0 += 64) {
    __syncthreads();
#pragma unroll
    for (int pass = 0; pass < 4; ++pass) {
      int idx = tid + pass * 256;
      int row = idx >> 3;          // 0..127
      int c8  = (idx & 7) << 3;    // 0..56
      *(v8bf*)&As[0][row * LDT + c8] = *(const v8bf*)&A [(size_t)(m0 + row) * K + k0 + c8];
      *(v8bf*)&Bs[0][row * LDT + c8] = *(const v8bf*)&Bw[(size_t)(n0 + row) * K + k0 + c8];
    }
    __syncthreads();
    if (k0 + 64 < K) {
      int prow = tid >> 1;
      __builtin_prefetch(&A [(size_t)(m0 + prow) * K + k0 + 64 + (tid & 1) * 32], 0, 1);
      __builtin_prefetch(&Bw[(size_t)(n0 + prow) * K + k0 + 64 + (tid & 1) * 32], 0, 1);
    }
#pragma unroll
    for (int ks = 0; ks < 64; ks += 32) {
      v16bf af[2], bfv[4];
#pragma unroll
      for (int i = 0; i < 2; ++i) af[i]  = load_a_frag_lds(&As[0][0], wm + i * 16, ks, LDT);
#pragma unroll
      for (int j = 0; j < 4; ++j) bfv[j] = load_b_frag_lds(&Bs[0][0], wn + j * 16, ks, LDT);
#pragma unroll
      for (int i = 0; i < 2; ++i)
#pragma unroll
        for (int j = 0; j < 4; ++j)
          acc[i][j] = __builtin_amdgcn_wmma_f32_16x16x32_bf16(
              false, af[i], false, bfv[j], (short)0, acc[i][j], false, false);
    }
  }
#endif

  // C/D layout: VGPR r -> row r + 8*(lane>=16), col = lane&15
  const int lane = tid & 31;
  const int colw = lane & 15;
  const int rowadd = (lane >> 4) << 3;
#pragma unroll
  for (int i = 0; i < 2; ++i)
#pragma unroll
    for (int j = 0; j < 4; ++j) {
      const int col = n0 + wn + j * 16 + colw;
      const float badd = ADD_BIAS ? bias[col] : 0.f;
#pragma unroll
      for (int r = 0; r < 8; ++r) {
        const int row = m0 + wm + i * 16 + rowadd + r;
        const float vv = acc[i][j][r] * out_scale + badd;
        if (WRITE_F32) ((float*)Cout)[(size_t)row * N + col] = vv;
        else           ((__bf16*)Cout)[(size_t)row * N + col] = (__bf16)vv;
      }
    }
}

// ---------------------------------------------------------------------------
// Causal flash attention. One workgroup per (b, h, 128-row Q tile).
// 8 waves, each owns 16 Q rows; Q fragments held in registers (scale folded
// into Q upstream). Iterates 64-key tiles: K tile DMA'd by the TDM with the
// LDS-pad feature producing the 136-elem padded stride directly (pad_interval
// =5 -> every 256B row, pad_amount=3 -> +16B); V transposed manually into LDS.
// S=Q·K^T via WMMA, online softmax (shuffle reductions match the C/D lane
// layout), P restaged through a per-wave LDS strip, O += P·V via WMMA.
// ---------------------------------------------------------------------------
__global__ __launch_bounds__(256) void attn_fa_kernel(
    const __bf16* __restrict__ Q, const __bf16* __restrict__ Km,
    const __bf16* __restrict__ V, __bf16* __restrict__ ctx)
{
  constexpr int T = 2048, D = 2048, HD = 128, QT = 128, KT = 64;
  __shared__ __bf16 Ks [KT * 136];   // 64 keys x 128 dims (+16B row pad)
  __shared__ __bf16 VTs[HD * 72];    // V^T: 128 dims x 64 keys
  __shared__ __bf16 Ps [QT * 72];    // P: 128 q-rows x 64 keys (per-wave strips)
  const int qt = blockIdx.x, h = blockIdx.y, b = blockIdx.z;
  const int qbase = qt * QT;
  const int tid = threadIdx.x, wid = tid >> 5, lane = tid & 31;
  const int colw = lane & 15, rowadd = (lane >> 4) << 3;
  const size_t headoff = (size_t)h * HD;

  // Q fragments (16 rows x 128 dims = 4 K-steps) resident in VGPRs
  v16bf aq[4];
  {
    const int qrow = qbase + wid * 16 + (lane & 15);
    const __bf16* qp = Q + ((size_t)(b * T + qrow)) * D + headoff;
    const int koff = (lane >> 4) << 3;
#pragma unroll
    for (int ks = 0; ks < 4; ++ks) {
      const __bf16* p = qp + ks * 32 + koff;
      ((v8bf*)&aq[ks])[0] = *(const v8bf*)p;
      ((v8bf*)&aq[ks])[1] = *(const v8bf*)(p + 16);
    }
  }

  float m_i[8], l_i[8];
  v8f oacc[8];
#pragma unroll
  for (int r = 0; r < 8; ++r) { m_i[r] = -1e30f; l_i[r] = 0.f; }
#pragma unroll
  for (int dt = 0; dt < 8; ++dt) oacc[dt] = v8f{};

  const int nkt = 2 * qt + 2;        // causal: only key tiles up to the diagonal
  for (int kt = 0; kt < nkt; ++kt) {
    const int kb = kt * KT;
    __syncthreads();                 // previous iteration done with Ks/VTs
#if USE_TDM
    if (wid == 0)                    // DMA K tile (64 rows x 128 dims, +pad)
      tdm_load_2d(lds_off32(&Ks[0]), Km + ((size_t)(b * T + kb)) * D + headoff,
                  128, 64, (unsigned)D, 128, 64,
                  0x00010000u | (1u << 20) | (5u << 22) | (3u << 25));
#endif
#pragma unroll
    for (int pass = 0; pass < 4; ++pass) {
      int idx = tid + pass * 256;
      int row = idx >> 4;            // 0..63  (key within tile)
      int c8  = (idx & 15) << 3;     // 0..120 (head dim chunk)
      const size_t g = ((size_t)(b * T + kb + row)) * D + headoff + c8;
#if !USE_TDM
      *(v8bf*)&Ks[row * 136 + c8] = *(const v8bf*)&Km[g];
#endif
      v8bf vv = *(const v8bf*)&V[g];
#pragma unroll
      for (int e = 0; e < 8; ++e) VTs[(c8 + e) * 72 + row] = vv[e];
    }
#if USE_TDM
    if (wid == 0) __builtin_amdgcn_s_wait_tensorcnt(0);
#endif
    __syncthreads();

    // S strip: 16 q-rows x 64 keys
    v8f sacc[4];
#pragma unroll
    for (int nt = 0; nt < 4; ++nt) {
      sacc[nt] = v8f{};
#pragma unroll
      for (int ks = 0; ks < 4; ++ks) {
        v16bf bk = load_b_frag_lds(Ks, nt * 16, ks * 32, 136);
        sacc[nt] = __builtin_amdgcn_wmma_f32_16x16x32_bf16(
            false, aq[ks], false, bk, (short)0, sacc[nt], false, false);
      }
    }
    // causal mask (only tiles that straddle the diagonal for this wave strip)
    if (kb + KT - 1 > qbase + wid * 16) {
#pragma unroll
      for (int nt = 0; nt < 4; ++nt) {
        const int colg = kb + nt * 16 + colw;
#pragma unroll
        for (int r = 0; r < 8; ++r) {
          const int rowg = qbase + wid * 16 + rowadd + r;
          if (colg > rowg) sacc[nt][r] = -__builtin_inff();
        }
      }
    }
    // online softmax: row-max / row-sum across the 16-lane column group
    float mnew[8], alpha[8], rsum[8];
#pragma unroll
    for (int r = 0; r < 8; ++r) {
      float v0 = sacc[0][r];
#pragma unroll
      for (int nt = 1; nt < 4; ++nt) v0 = fmaxf(v0, sacc[nt][r]);
      mnew[r] = v0;
    }
#pragma unroll
    for (int off = 1; off < 16; off <<= 1)
#pragma unroll
      for (int r = 0; r < 8; ++r) mnew[r] = fmaxf(mnew[r], __shfl_xor(mnew[r], off, 32));
#pragma unroll
    for (int r = 0; r < 8; ++r) {
      mnew[r]  = fmaxf(mnew[r], m_i[r]);
      alpha[r] = __expf(m_i[r] - mnew[r]);
      m_i[r]   = mnew[r];
      rsum[r]  = 0.f;
    }
#pragma unroll
    for (int nt = 0; nt < 4; ++nt)
#pragma unroll
      for (int r = 0; r < 8; ++r) {
        const float p = __expf(sacc[nt][r] - mnew[r]);
        sacc[nt][r] = p;
        rsum[r] += p;
      }
#pragma unroll
    for (int off = 1; off < 16; off <<= 1)
#pragma unroll
      for (int r = 0; r < 8; ++r) rsum[r] += __shfl_xor(rsum[r], off, 32);
#pragma unroll
    for (int r = 0; r < 8; ++r) l_i[r] = l_i[r] * alpha[r] + rsum[r];
#pragma unroll
    for (int dt = 0; dt < 8; ++dt)
#pragma unroll
      for (int r = 0; r < 8; ++r) oacc[dt][r] *= alpha[r];

    // P (C/D layout) -> per-wave LDS strip -> A-fragment layout
#pragma unroll
    for (int nt = 0; nt < 4; ++nt)
#pragma unroll
      for (int r = 0; r < 8; ++r)
        Ps[(wid * 16 + rowadd + r) * 72 + nt * 16 + colw] = (__bf16)sacc[nt][r];

    // O += P · V  (K-dim = 64 keys, N-dim = 128 head dims via V^T rows)
#pragma unroll
    for (int js = 0; js < 2; ++js) {
      v16bf ap = load_a_frag_lds(Ps, wid * 16, js * 32, 72);
#pragma unroll
      for (int dt = 0; dt < 8; ++dt) {
        v16bf bv = load_b_frag_lds(VTs, dt * 16, js * 32, 72);
        oacc[dt] = __builtin_amdgcn_wmma_f32_16x16x32_bf16(
            false, ap, false, bv, (short)0, oacc[dt], false, false);
      }
    }
  }

  // normalize and write context
  float inv_l[8];
#pragma unroll
  for (int r = 0; r < 8; ++r) inv_l[r] = 1.0f / l_i[r];
#pragma unroll
  for (int dt = 0; dt < 8; ++dt) {
    const int colg = dt * 16 + colw;
#pragma unroll
    for (int r = 0; r < 8; ++r) {
      const int rowg = qbase + wid * 16 + rowadd + r;
      ctx[((size_t)(b * T + rowg)) * D + headoff + colg] = (__bf16)(oacc[dt][r] * inv_l[r]);
    }
  }
}

// ---------------------------------------------------------------------------
extern "C" void kernel_launch(void* const* d_in, const int* in_sizes, int n_in,
                              void* d_out, int out_size, void* d_ws, size_t ws_size,
                              hipStream_t stream) {
  (void)in_sizes; (void)n_in; (void)out_size; (void)ws_size;
  const float* x  = (const float*)d_in[0];
  const float* Wq = (const float*)d_in[1];
  const float* Wk = (const float*)d_in[2];
  const float* Wv = (const float*)d_in[3];
  const float* Wo = (const float*)d_in[4];
  const float* bo = (const float*)d_in[5];

  constexpr int B = 2, T = 2048, D = 2048;
  constexpr size_t NTOK = (size_t)B * T;   // 4096
  constexpr size_t XE = NTOK * D;          // 8M elems
  constexpr size_t WE = (size_t)D * D;     // 4M elems

  char* w = (char*)d_ws;
  __bf16* xb  = (__bf16*)w; w += XE * 2;
  __bf16* wqb = (__bf16*)w; w += WE * 2;
  __bf16* wkb = (__bf16*)w; w += WE * 2;
  __bf16* wvb = (__bf16*)w; w += WE * 2;
  __bf16* wob = (__bf16*)w; w += WE * 2;
  __bf16* qb  = (__bf16*)w; w += XE * 2;
  __bf16* kbp = (__bf16*)w; w += XE * 2;
  __bf16* vbp = (__bf16*)w; w += XE * 2;
  __bf16* cb  = (__bf16*)w; w += XE * 2;

  f32_to_bf16_kernel<<<1024, 256, 0, stream>>>(x,  xb,  XE / 4);
  f32_to_bf16_kernel<<<512,  256, 0, stream>>>(Wq, wqb, WE / 4);
  f32_to_bf16_kernel<<<512,  256, 0, stream>>>(Wk, wkb, WE / 4);
  f32_to_bf16_kernel<<<512,  256, 0, stream>>>(Wv, wvb, WE / 4);
  f32_to_bf16_kernel<<<512,  256, 0, stream>>>(Wo, wob, WE / 4);

  const dim3 gg((unsigned)(NTOK / 128), (unsigned)(D / 128));   // 32 x 16
  const float qscale = 0.088388347648318447f;                   // 1/sqrt(HD=128)
  gemm_bf16_wmma<false,false><<<gg, 256, 0, stream>>>(xb, wqb, nullptr, qb,  (int)NTOK, D, D, qscale);
  gemm_bf16_wmma<false,false><<<gg, 256, 0, stream>>>(xb, wkb, nullptr, kbp, (int)NTOK, D, D, 1.0f);
  gemm_bf16_wmma<false,false><<<gg, 256, 0, stream>>>(xb, wvb, nullptr, vbp, (int)NTOK, D, D, 1.0f);

  attn_fa_kernel<<<dim3(T / 128, 16, B), 256, 0, stream>>>(qb, kbp, vbp, cb);

  gemm_bf16_wmma<true,true><<<gg, 256, 0, stream>>>(cb, wob, bo, d_out, (int)NTOK, D, D, 1.0f);
}
